// TransformerDecoderBlock_30923764531643
// MI455X (gfx1250) — compile-verified, matrix operand order checked by reading, not run on hardware
//
#include <hip/hip_runtime.h>
#include <hip/hip_bf16.h>

// ---------------------------------------------------------------------------
// Types for CDNA5 WMMA (wave32): v_wmma_f32_16x16x32_bf16
// ---------------------------------------------------------------------------
typedef __attribute__((ext_vector_type(8)))  __bf16 v8bf;
typedef __attribute__((ext_vector_type(16))) __bf16 v16bf;
typedef __attribute__((ext_vector_type(8)))  float  v8f;
typedef __attribute__((ext_vector_type(4)))  int    v4i;

typedef __attribute__((address_space(1))) void gvoid_t;
typedef __attribute__((address_space(3))) void lvoid_t;
typedef __attribute__((address_space(1))) v4i  gv4i_t;
typedef __attribute__((address_space(3))) v4i  lv4i_t;

// ---------------------------------------------------------------------------
// Async global->LDS copy (CDNA5 GLOBAL_LOAD_ASYNC_TO_LDS_B128, ASYNCcnt).
// Probe-confirmed signature: (int4 AS1*, int4 AS3*, imm offset, imm cpol).
// ---------------------------------------------------------------------------
static __device__ inline void async_cp_b128(const void* g, void* l) {
#if __has_builtin(__builtin_amdgcn_global_load_async_to_lds_b128)
  __builtin_amdgcn_global_load_async_to_lds_b128(
      (gv4i_t*)(gvoid_t*)(void*)g, (lv4i_t*)(lvoid_t*)l, 0, 0);
#else
  unsigned loff = (unsigned)(unsigned long long)(lvoid_t*)l;
  asm volatile("global_load_async_to_lds_b128 %0, %1, off"
               :: "v"(loff), "v"(g) : "memory");
#endif
}

static __device__ inline void async_wait0() {
#if __has_builtin(__builtin_amdgcn_s_wait_asynccnt)
  __builtin_amdgcn_s_wait_asynccnt(0);
#else
  asm volatile("s_wait_asynccnt 0x0" ::: "memory");
#endif
}

// Load a 16-bit A/B fragment (16x32, n-major rows of 32 K) from a row base.
// Per ISA: lanes 0-15 hold K{0..7,16..23}, lanes 16-31 hold K{8..15,24..31}.
static __device__ inline v16bf ld_frag(const __bf16* rowbase, int halfsel) {
  v8bf lo = *(const v8bf*)(rowbase + halfsel * 8);
  v8bf hi = *(const v8bf*)(rowbase + 16 + halfsel * 8);
  v16bf r;
#pragma unroll
  for (int i = 0; i < 8; ++i) { r[i] = lo[i]; r[i + 8] = hi[i]; }
  return r;
}

// ---------------------------------------------------------------------------
// Elementwise conversion + weight repack kernels (run once per call; cheap)
// ---------------------------------------------------------------------------
__global__ void f32_to_bf16_kernel(const float* __restrict__ in,
                                   __bf16* __restrict__ out, int n) {
  int i = blockIdx.x * blockDim.x + threadIdx.x;
  if (i < n) out[i] = (__bf16)in[i];
}

// W [H=16][K=1024][E=64] (per-head proj) -> out [N=1024][K=1024] (B^T layout)
__global__ void repack_headed_kernel(const float* __restrict__ W,
                                     __bf16* __restrict__ out) {
  int i = blockIdx.x * blockDim.x + threadIdx.x;  // i = n*1024 + k
  int n = i >> 10, k = i & 1023;
  int h = n >> 6, e = n & 63;
  out[i] = (__bf16)W[(size_t)h * 65536 + (size_t)k * 64 + e];
}

// W [K][N] row-major -> out [N][K] bf16 (B^T layout)
__global__ void transpose_to_bf16_kernel(const float* __restrict__ W,
                                         __bf16* __restrict__ out,
                                         int K, int N) {
  int i = blockIdx.x * blockDim.x + threadIdx.x;
  if (i >= K * N) return;
  int n = i / K, k = i - n * K;
  out[i] = (__bf16)W[(size_t)k * N + n];
}

// ---------------------------------------------------------------------------
// bf16 WMMA GEMM: C[M,N] = A[M,K] * Bt[N,K]^T (+bias, optional relu)
// 128 threads (4 waves), block tile 128x128, wave tile 64x64 (16 WMMA/step),
// K step 32, double-buffered LDS fed by async global->LDS copies,
// one barrier per K step.
// ---------------------------------------------------------------------------
#define GSTR 40

__global__ __launch_bounds__(128) void gemm_bf16_kernel(
    const __bf16* __restrict__ A, const __bf16* __restrict__ Bt,
    const float* __restrict__ bias,
    float* __restrict__ Cf, __bf16* __restrict__ Cb,
    int M, int N, int K, int relu) {
  __shared__ __bf16 a_s[2][128 * GSTR];
  __shared__ __bf16 b_s[2][128 * GSTR];

  const int tid  = threadIdx.x;
  const int lane = tid & 31;
  const int w    = tid >> 5;         // 4 waves
  const int wm   = (w & 1) * 64;     // 2 waves along M
  const int wn   = (w >> 1) * 64;    // 2 waves along N
  const int mblk = blockIdx.y * 128;
  const int nblk = blockIdx.x * 128;
  const int hs   = lane >> 4;
  const int l16  = lane & 15;

  // Per-thread staging chunks: 4 for A + 4 for B (each tile 128x32 bf16).
  int srow[4], soff[4];
#pragma unroll
  for (int c = 0; c < 4; ++c) {
    int ch = tid + c * 128;
    srow[c] = ch >> 2;
    soff[c] = (ch & 3) * 8;
  }

  auto stage = [&](int buf, int k0) {
#pragma unroll
    for (int c = 0; c < 4; ++c) {
      async_cp_b128(&A[(size_t)(mblk + srow[c]) * K + k0 + soff[c]],
                    &a_s[buf][srow[c] * GSTR + soff[c]]);
      async_cp_b128(&Bt[(size_t)(nblk + srow[c]) * K + k0 + soff[c]],
                    &b_s[buf][srow[c] * GSTR + soff[c]]);
    }
  };

  v8f zero = {};
  v8f acc[4][4];
#pragma unroll
  for (int i = 0; i < 4; ++i)
#pragma unroll
    for (int j = 0; j < 4; ++j) acc[i][j] = zero;

  stage(0, 0);
  async_wait0();
  __syncthreads();

  int cur = 0;
  for (int k0 = 0; k0 < K; k0 += 32) {
    const bool more = (k0 + 32 < K);
    if (more) stage(cur ^ 1, k0 + 32);       // stream next slab into other buf
    if (k0 + 64 < K) {                       // GL2 hint for slab after next
      __builtin_prefetch(&A[(size_t)(mblk + srow[0]) * K + k0 + 64], 0, 2);
      __builtin_prefetch(&Bt[(size_t)(nblk + srow[0]) * K + k0 + 64], 0, 2);
    }

    v16bf af[4], bfr[4];
#pragma unroll
    for (int i = 0; i < 4; ++i)
      af[i] = ld_frag(&a_s[cur][(wm + i * 16 + l16) * GSTR], hs);
#pragma unroll
    for (int j = 0; j < 4; ++j)
      bfr[j] = ld_frag(&b_s[cur][(wn + j * 16 + l16) * GSTR], hs);

#pragma unroll
    for (int i = 0; i < 4; ++i)
#pragma unroll
      for (int j = 0; j < 4; ++j)
        acc[i][j] = __builtin_amdgcn_wmma_f32_16x16x32_bf16(
            false, af[i], false, bfr[j], (short)0, acc[i][j], false, false);

    if (more) async_wait0();
    __syncthreads();
    cur ^= 1;
  }

  // Epilogue: bias + optional relu, write f32 or bf16.
#pragma unroll
  for (int i = 0; i < 4; ++i) {
#pragma unroll
    for (int j = 0; j < 4; ++j) {
      int n = nblk + wn + j * 16 + l16;
      float bv = bias ? bias[n] : 0.f;
#pragma unroll
      for (int r = 0; r < 8; ++r) {
        int m = mblk + wm + i * 16 + hs * 8 + r;
        float v = acc[i][j][r] + bv;
        if (relu) v = fmaxf(v, 0.f);
        if (Cb) Cb[(size_t)m * N + n] = (__bf16)v;
        else    Cf[(size_t)m * N + n] = v;
      }
    }
  }
}

// ---------------------------------------------------------------------------
// Flash attention (bf16 WMMA, f32 softmax state). One wave = 16 query rows.
// Block = 8 waves = 128 query rows, shares K / V^T tiles (32 keys) in LDS.
// K tile staged via async global->LDS; V^T via scatter-transpose.
// ---------------------------------------------------------------------------
#define KSTR 72
#define VSTR 40
#define PSTR 40

__global__ __launch_bounds__(256) void attn_kernel(
    const __bf16* __restrict__ Q, const __bf16* __restrict__ Kb,
    const __bf16* __restrict__ Vb, float* __restrict__ O,
    int S, int T, int causal) {
  __shared__ __bf16 k_s[32 * KSTR];        // K tile: 32 keys x 64 dims
  __shared__ __bf16 vt_s[64 * VSTR];       // V^T tile: 64 dims x 32 keys
  __shared__ __bf16 p_s[8 * 16 * PSTR];    // per-wave P transpose scratch

  const int tid  = threadIdx.x;
  const int lane = tid & 31;
  const int w    = tid >> 5;
  const int hs   = lane >> 4;
  const int l16  = lane & 15;
  const int h    = blockIdx.y;
  const int b    = blockIdx.z;
  const int q0   = blockIdx.x * 128 + w * 16;
  const int D    = 1024;
  const float scale = 0.125f;  // 64^-0.5

  // Q fragments held in registers for the whole pass (2 chunks of K=32).
  const __bf16* qrow = Q + (size_t)(b * S + q0 + l16) * D + h * 64;
  v16bf qf[2];
  qf[0] = ld_frag(qrow, hs);
  qf[1] = ld_frag(qrow + 32, hs);

  v8f zero = {};
  float mi[8], li[8];
  v8f o[4];
#pragma unroll
  for (int r = 0; r < 8; ++r) { mi[r] = -3.0e38f; li[r] = 0.f; }
#pragma unroll
  for (int j = 0; j < 4; ++j) o[j] = zero;

  __bf16* pw = &p_s[w * 16 * PSTR];

  const int tend = causal ? ((int)blockIdx.x * 128 + 128) : T;
  const int Tmax = tend < T ? tend : T;

  for (int t0 = 0; t0 < Tmax; t0 += 32) {
    {
      int row = tid >> 3, off = (tid & 7) * 8;
      // K tile: direct async DMA into LDS (row-major == WMMA-B compatible).
      async_cp_b128(&Kb[(size_t)(b * T + t0 + row) * D + h * 64 + off],
                    &k_s[row * KSTR + off]);
      // V^T tile: load + scatter-transpose (8 b16 LDS stores).
      v8bf vv = *(const v8bf*)&Vb[(size_t)(b * T + t0 + row) * D + h * 64 + off];
#pragma unroll
      for (int j = 0; j < 8; ++j) vt_s[(off + j) * VSTR + row] = vv[j];
    }
    async_wait0();
    __syncthreads();

    // Scores for two 16-key groups: S = Q * K^T (K depth 64 = 2 WMMAs each).
    v8f sc[2];
    sc[0] = zero; sc[1] = zero;
#pragma unroll
    for (int kg = 0; kg < 2; ++kg) {
      const __bf16* krow = &k_s[(kg * 16 + l16) * KSTR];
#pragma unroll
      for (int c = 0; c < 2; ++c) {
        v16bf kf = ld_frag(krow + c * 32, hs);
        sc[kg] = __builtin_amdgcn_wmma_f32_16x16x32_bf16(
            false, qf[c], false, kf, (short)0, sc[kg], false, false);
      }
    }

    // Scale + causal mask. C layout: element (m = r+8*hs, n = l16).
#pragma unroll
    for (int kg = 0; kg < 2; ++kg) {
      int kid = t0 + kg * 16 + l16;
#pragma unroll
      for (int r = 0; r < 8; ++r) {
        float s = sc[kg][r] * scale;
        int qid = q0 + hs * 8 + r;
        if (causal && kid > qid) s = -3.0e38f;
        sc[kg][r] = s;
      }
    }

    // Online softmax per row; row lives across 16 lanes of the same half.
#pragma unroll
    for (int r = 0; r < 8; ++r) {
      float rm = fmaxf(sc[0][r], sc[1][r]);
#pragma unroll
      for (int x = 1; x < 16; x <<= 1) rm = fmaxf(rm, __shfl_xor(rm, x, 32));
      float mnew = fmaxf(mi[r], rm);
      float corr = __expf(mi[r] - mnew);
      float p0 = __expf(sc[0][r] - mnew);
      float p1 = __expf(sc[1][r] - mnew);
      float rs = p0 + p1;
#pragma unroll
      for (int x = 1; x < 16; x <<= 1) rs += __shfl_xor(rs, x, 32);
      li[r] = li[r] * corr + rs;
      mi[r] = mnew;
#pragma unroll
      for (int j = 0; j < 4; ++j) o[j][r] *= corr;
      int m = hs * 8 + r;
      // Wave-private LDS transpose C-layout -> A-layout (in-order LDS, no bar)
      pw[m * PSTR + l16]      = (__bf16)p0;
      pw[m * PSTR + 16 + l16] = (__bf16)p1;
    }

    // O += P[16x32] * V[32x64]  (4 WMMAs over head-dim tiles).
    v16bf pf = ld_frag(&pw[l16 * PSTR], hs);
#pragma unroll
    for (int j = 0; j < 4; ++j) {
      v16bf vf = ld_frag(&vt_s[(j * 16 + l16) * VSTR], hs);
      o[j] = __builtin_amdgcn_wmma_f32_16x16x32_bf16(
          false, pf, false, vf, (short)0, o[j], false, false);
    }
    __syncthreads();
  }

  // Normalize and write f32 output at [b, q, h*64 + d].
#pragma unroll
  for (int r = 0; r < 8; ++r) {
    float inv = 1.0f / li[r];
    int m = q0 + hs * 8 + r;
#pragma unroll
    for (int j = 0; j < 4; ++j)
      O[(size_t)(b * S + m) * D + h * 64 + j * 16 + l16] = o[j][r] * inv;
  }
}

// ---------------------------------------------------------------------------
// Fused residual add + LayerNorm. One block (256 thr) per 1024-wide row.
// Emits f32 residual stream and (optionally) bf16 copy for the next GEMM.
// ---------------------------------------------------------------------------
__global__ __launch_bounds__(256) void add_ln_kernel(
    const float* __restrict__ X, const float* __restrict__ Y,
    const float* __restrict__ g, const float* __restrict__ be,
    float* __restrict__ outf, __bf16* __restrict__ outb) {
  __shared__ float red[16];
  const int row = blockIdx.x;
  const int tid = threadIdx.x;
  const int D = 1024;
  float v[4];
  float s = 0.f, s2 = 0.f;
#pragma unroll
  for (int i = 0; i < 4; ++i) {
    int c = tid + i * 256;
    float x = X[(size_t)row * D + c] + Y[(size_t)row * D + c];
    v[i] = x; s += x; s2 += x * x;
  }
#pragma unroll
  for (int x = 1; x < 32; x <<= 1) {
    s  += __shfl_xor(s,  x, 32);
    s2 += __shfl_xor(s2, x, 32);
  }
  int wv = tid >> 5, lane = tid & 31;
  if (lane == 0) { red[wv] = s; red[8 + wv] = s2; }
  __syncthreads();
  if (wv == 0) {
    float a  = (lane < 8) ? red[lane] : 0.f;
    float b2 = (lane < 8) ? red[8 + lane] : 0.f;
#pragma unroll
    for (int x = 1; x < 8; x <<= 1) {
      a  += __shfl_xor(a,  x, 32);
      b2 += __shfl_xor(b2, x, 32);
    }
    if (lane == 0) { red[0] = a; red[1] = b2; }
  }
  __syncthreads();
  float mean = red[0] * (1.0f / 1024.0f);
  float var  = red[1] * (1.0f / 1024.0f) - mean * mean;
  float rstd = rsqrtf(var + 1e-5f);
#pragma unroll
  for (int i = 0; i < 4; ++i) {
    int c = tid + i * 256;
    float y = (v[i] - mean) * rstd * g[c] + be[c];
    if (outf) outf[(size_t)row * D + c] = y;
    if (outb) outb[(size_t)row * D + c] = (__bf16)y;
  }
}

// ---------------------------------------------------------------------------
// Host-side orchestration
// ---------------------------------------------------------------------------
extern "C" void kernel_launch(void* const* d_in, const int* in_sizes, int n_in,
                              void* d_out, int out_size, void* d_ws, size_t ws_size,
                              hipStream_t stream) {
  const int B = 2, S = 2048, T = 2048, Dm = 1024, FF = 4096;
  const int BS = B * S;  // 4096 rows

  const float* in_x   = (const float*)d_in[0];
  const float* in_enc = (const float*)d_in[1];
  const float* Wq1 = (const float*)d_in[2];
  const float* Wk1 = (const float*)d_in[3];
  const float* Wv1 = (const float*)d_in[4];
  const float* bv1 = (const float*)d_in[5];
  const float* Wq2 = (const float*)d_in[6];
  const float* Wk2 = (const float*)d_in[7];
  const float* Wv2 = (const float*)d_in[8];
  const float* bv2 = (const float*)d_in[9];
  const float* g1  = (const float*)d_in[10];
  const float* be1 = (const float*)d_in[11];
  const float* g2  = (const float*)d_in[12];
  const float* be2 = (const float*)d_in[13];
  const float* g3  = (const float*)d_in[14];
  const float* be3 = (const float*)d_in[15];
  const float* Wf1 = (const float*)d_in[16];
  const float* bf1 = (const float*)d_in[17];
  const float* Wf2 = (const float*)d_in[18];
  const float* bf2 = (const float*)d_in[19];

  char* ws = (char*)d_ws;
  size_t off = 0;
  auto alloc = [&](size_t bytes) -> void* {
    void* p = ws + off;
    off = (off + bytes + 255) & ~(size_t)255;
    return p;
  };

  __bf16* xb   = (__bf16*)alloc((size_t)BS * Dm * 2);
  __bf16* encb = (__bf16*)alloc((size_t)B * T * Dm * 2);
  __bf16* wq1b = (__bf16*)alloc((size_t)Dm * Dm * 2);
  __bf16* wk1b = (__bf16*)alloc((size_t)Dm * Dm * 2);
  __bf16* wv1b = (__bf16*)alloc((size_t)Dm * Dm * 2);
  __bf16* wq2b = (__bf16*)alloc((size_t)Dm * Dm * 2);
  __bf16* wk2b = (__bf16*)alloc((size_t)Dm * Dm * 2);
  __bf16* wv2b = (__bf16*)alloc((size_t)Dm * Dm * 2);
  __bf16* wf1b = (__bf16*)alloc((size_t)Dm * FF * 2);
  __bf16* wf2b = (__bf16*)alloc((size_t)Dm * FF * 2);
  __bf16* qb   = (__bf16*)alloc((size_t)BS * Dm * 2);
  __bf16* kb   = (__bf16*)alloc((size_t)B * T * Dm * 2);
  __bf16* vb   = (__bf16*)alloc((size_t)B * T * Dm * 2);
  float*  attnf= (float*)alloc((size_t)BS * Dm * 4);
  float*  x1f  = (float*)alloc((size_t)BS * Dm * 4);
  __bf16* hb   = (__bf16*)alloc((size_t)BS * FF * 2);
  (void)ws_size;

  // 1. Activations to bf16.
  {
    int n = BS * Dm;
    f32_to_bf16_kernel<<<(n + 255) / 256, 256, 0, stream>>>(in_x, xb, n);
    f32_to_bf16_kernel<<<(n + 255) / 256, 256, 0, stream>>>(in_enc, encb, n);
  }
  // 2. Repack weights to bf16 B^T layout.
  {
    int n = Dm * Dm;
    repack_headed_kernel<<<n / 256, 256, 0, stream>>>(Wq1, wq1b);
    repack_headed_kernel<<<n / 256, 256, 0, stream>>>(Wk1, wk1b);
    repack_headed_kernel<<<n / 256, 256, 0, stream>>>(Wv1, wv1b);
    repack_headed_kernel<<<n / 256, 256, 0, stream>>>(Wq2, wq2b);
    repack_headed_kernel<<<n / 256, 256, 0, stream>>>(Wk2, wk2b);
    repack_headed_kernel<<<n / 256, 256, 0, stream>>>(Wv2, wv2b);
    int nf = Dm * FF;
    transpose_to_bf16_kernel<<<nf / 256, 256, 0, stream>>>(Wf1, wf1b, Dm, FF);
    transpose_to_bf16_kernel<<<nf / 256, 256, 0, stream>>>(Wf2, wf2b, FF, Dm);
  }

  dim3 blk(128);
  dim3 gproj(Dm / 128, BS / 128);
  // 3. Self-attention projections.
  gemm_bf16_kernel<<<gproj, blk, 0, stream>>>(xb, wq1b, nullptr, nullptr, qb, BS, Dm, Dm, 0);
  gemm_bf16_kernel<<<gproj, blk, 0, stream>>>(xb, wk1b, nullptr, nullptr, kb, BS, Dm, Dm, 0);
  gemm_bf16_kernel<<<gproj, blk, 0, stream>>>(xb, wv1b, bv1,     nullptr, vb, BS, Dm, Dm, 0);
  // 4. Causal flash attention.
  attn_kernel<<<dim3(S / 128, 16, B), dim3(256), 0, stream>>>(qb, kb, vb, attnf, S, T, 1);
  // 5. x1 = LN(attn + input); xb = bf16(x1).
  add_ln_kernel<<<BS, dim3(256), 0, stream>>>(attnf, in_x, g1, be1, x1f, xb);
  // 6. Cross-attention projections.
  gemm_bf16_kernel<<<gproj, blk, 0, stream>>>(xb,   wq2b, nullptr, nullptr, qb, BS, Dm, Dm, 0);
  gemm_bf16_kernel<<<gproj, blk, 0, stream>>>(encb, wk2b, nullptr, nullptr, kb, BS, Dm, Dm, 0);
  gemm_bf16_kernel<<<gproj, blk, 0, stream>>>(encb, wv2b, bv2,     nullptr, vb, BS, Dm, Dm, 0);
  // 7. Non-causal flash attention.
  attn_kernel<<<dim3(S / 128, 16, B), dim3(256), 0, stream>>>(qb, kb, vb, attnf, S, T, 0);
  // 8. x1 = LN(x1 + cross_attn); xb = bf16(x1).
  add_ln_kernel<<<BS, dim3(256), 0, stream>>>(x1f, attnf, g2, be2, x1f, xb);
  // 9. FFN: h = relu(x @ Wf1 + bf1) (bf16 out).
  gemm_bf16_kernel<<<dim3(FF / 128, BS / 128), blk, 0, stream>>>(
      xb, wf1b, bf1, nullptr, hb, BS, FF, Dm, 1);
  // 10. ffn_out = h @ Wf2 + bf2 (f32 out, reuse attnf).
  gemm_bf16_kernel<<<dim3(Dm / 128, BS / 128), blk, 0, stream>>>(
      hb, wf2b, bf2, attnf, nullptr, BS, Dm, FF, 0);
  // 11. out = LN(x1 + ffn_out).
  add_ln_kernel<<<BS, dim3(256), 0, stream>>>(x1f, attnf, g3, be3, (float*)d_out, nullptr);
}
